// SSR_wo_conv_65412351918428
// MI455X (gfx1250) — matrix operand order repaired and checked by reading, not compiled, not run on hardware
//
#include <hip/hip_runtime.h>
#include <hip/hip_bf16.h>
#include <math.h>

// ---------------------------------------------------------------------------
// CDNA5 (gfx1250): bf16 WMMA 16x16x32 (f32 accum) + async global->LDS staging
// ---------------------------------------------------------------------------

typedef __attribute__((ext_vector_type(16))) __bf16 v16bf;
typedef __attribute__((ext_vector_type(8)))  float  v8f;

union RegTile { unsigned int u[8]; v16bf v; };

#ifndef __has_builtin
#define __has_builtin(x) 0
#endif
#if __has_builtin(__builtin_amdgcn_global_load_async_to_lds_b128) && \
    __has_builtin(__builtin_amdgcn_s_wait_asynccnt)
#define HAVE_ASYNC_LDS 1
typedef __attribute__((ext_vector_type(4))) int v4i;
typedef __attribute__((address_space(1))) v4i as1_v4i;   // global int4
typedef __attribute__((address_space(3))) v4i as3_v4i;   // LDS int4
#else
#define HAVE_ASYNC_LDS 0
#endif

#define NT       65536      // total tokens (window-major order)
#define EMB      180
#define EPAD     192
#define NWIN     256
#define NHEAD    6
#define HD       30
#define HDP      32
#define QKV_NP   576        // 540 padded to 18*32 (and 9*64)
#define MLP_H    360
#define MLP_HP   384
#define UP_NP    64         // 48 padded

// A-tile K offset for element-pair j, lane-half kh (16-bit A, 16x32)
#define AK(j,kh) ((((j) < 4) ? ((j)*2) : (16 + ((j)-4)*2)) + ((kh)*8))

__device__ __forceinline__ float gelu_tanh(float x) {
  const float k0 = 0.7978845608028654f, k1 = 0.044715f;
  return 0.5f * x * (1.f + tanhf(k0 * (x + k1 * x * x * x)));
}

__device__ __forceinline__ float cubw(float d) {  // Keys cubic, a = -0.5
  d = fabsf(d);
  if (d <= 1.f) return ((1.5f * d - 2.5f) * d) * d + 1.f;
  if (d <  2.f) return ((-0.5f * d + 2.5f) * d - 4.f) * d + 2.f;
  return 0.f;
}

// ----------------- weight pad/convert, TRANSPOSED: dst[n][k] ---------------
__global__ void convert_padT_kernel(const float* __restrict__ src, __bf16* __restrict__ dst,
                                    int K, int N, int Kp, int Np) {
  int idx = blockIdx.x * 256 + threadIdx.x;   // over Np*Kp
  if (idx >= Np * Kp) return;
  int n = idx / Kp, k = idx - n * Kp;
  dst[idx] = (k < K && n < N) ? (__bf16)src[k * N + n] : (__bf16)0.f;
}

// --------------------------- selector + gumbel -----------------------------
__global__ void sel_kernel(const float* __restrict__ x, const float* __restrict__ u,
                           const float* s1w, const float* s1b,
                           const float* s2w, const float* s2b,
                           const float* s3w, const float* s3b,
                           float* mask, float* out_f3, float* out_f2, float* out_f1) {
  __shared__ float pooled[192];
  __shared__ float r1[64];
  __shared__ float r2[32];
  __shared__ float f3s[2];
  int tile = blockIdx.x, j = threadIdx.x;
  int ty = tile >> 2, tx = tile & 3;
  {
    int py = j >> 3, px = j & 7;
    for (int c = 0; c < 3; ++c) {
      float s = 0.f;
      for (int iy = 0; iy < 8; ++iy)
        for (int ix = 0; ix < 8; ++ix) {
          int Y = ty * 64 + py * 8 + iy, X = tx * 64 + px * 8 + ix;
          s += x[c * 65536 + Y * 256 + X];
        }
      pooled[(py * 8 + px) * 3 + c] = s * (1.f / 64.f);
    }
  }
  __syncthreads();
  {
    float a = s1b[j];
    for (int k = 0; k < 192; ++k) a += pooled[k] * s1w[k * 64 + j];
    out_f1[tile * 64 + j] = a;
    r1[j] = fmaxf(a, 0.f);
  }
  __syncthreads();
  if (j < 32) {
    float a = s2b[j];
    for (int k = 0; k < 64; ++k) a += r1[k] * s2w[k * 32 + j];
    out_f2[tile * 32 + j] = a;
    r2[j] = fmaxf(a, 0.f);
  }
  __syncthreads();
  if (j < 2) {
    float a = s3b[j];
    for (int k = 0; k < 32; ++k) a += r2[k] * s3w[k * 2 + j];
    out_f3[tile * 2 + j] = a;
    f3s[j] = a;
  }
  __syncthreads();
  if (j == 0) {
    float g0 = -logf(-logf(u[tile * 2 + 0] + 1e-10f) + 1e-10f);
    float g1 = -logf(-logf(u[tile * 2 + 1] + 1e-10f) + 1e-10f);
    mask[tile] = ((f3s[1] + g1) > (f3s[0] + g0)) ? 1.f : 0.f;
  }
}

// ------------------------------- embedding ---------------------------------
__global__ __launch_bounds__(256) void embed_kernel(const float* __restrict__ x,
                                                    const float* __restrict__ ew,
                                                    const float* __restrict__ eb,
                                                    float* __restrict__ tok) {
  int idx = blockIdx.x * 256 + threadIdx.x;   // NT*EMB
  if (idx >= NT * EMB) return;
  int t = idx / EMB, e = idx - t * EMB;
  int g = t >> 8, l = t & 255;
  int tile = g >> 4, w = g & 15;
  int Y = (tile >> 2) * 64 + (w >> 2) * 16 + (l >> 4);
  int X = (tile & 3) * 64 + (w & 3) * 16 + (l & 15);
  float p0 = x[0 * 65536 + Y * 256 + X];
  float p1 = x[1 * 65536 + Y * 256 + X];
  float p2 = x[2 * 65536 + Y * 256 + X];
  tok[idx] = p0 * ew[0 * EMB + e] + p1 * ew[1 * EMB + e] + p2 * ew[2 * EMB + e] + eb[e];
}

// ------------------- layernorm (or plain convert) -> bf16 ------------------
__global__ __launch_bounds__(256) void ln_kernel(const float* __restrict__ in,
                                                 __bf16* __restrict__ out,
                                                 const float* __restrict__ gw,
                                                 const float* __restrict__ bw, int doLN) {
  __shared__ float red[256];
  int t = blockIdx.x, e = threadIdx.x;
  float x = (e < EMB) ? in[(size_t)t * EMB + e] : 0.f;
  float y = x;
  if (doLN) {
    red[e] = x;  __syncthreads();
    for (int s = 128; s > 0; s >>= 1) { if (e < s) red[e] += red[e + s]; __syncthreads(); }
    float mu = red[0] * (1.f / EMB);
    __syncthreads();
    float d = (e < EMB) ? x - mu : 0.f;
    red[e] = d * d;  __syncthreads();
    for (int s = 128; s > 0; s >>= 1) { if (e < s) red[e] += red[e + s]; __syncthreads(); }
    float var = red[0] * (1.f / EMB);
    y = (e < EMB) ? d * rsqrtf(var + 1e-5f) * gw[e] + bw[e] : 0.f;
  }
  if (e < EPAD) out[(size_t)t * EPAD + e] = (e < EMB) ? (__bf16)y : (__bf16)0.f;
}

// ------------------------- generic bf16 WMMA GEMM --------------------------
// C[M x N] = A[M x K] * B[K x N], with B given TRANSPOSED: BT[N][K].
// M%128==0, N%64==0, K%32==0. 8 waves: 128x64 block tile, 4 WMMAs/wave/step.
// Double-buffered LDS panels filled by GLOBAL_LOAD_ASYNC_TO_LDS_B128 when
// available (ASYNCcnt pipeline), synchronous b128 staging otherwise.
__global__ __launch_bounds__(256) void gemm_bf16_kernel(const __bf16* __restrict__ A,
                                                        const __bf16* __restrict__ BT,
                                                        float* __restrict__ C,
                                                        int M, int N, int K) {
  __shared__ __bf16 As[2][128 * 32];
  __shared__ __bf16 Bs[2][64 * 32];   // Bs[n][k]
  int m0 = blockIdx.x * 128;
  int n0 = blockIdx.y * 64;
  int tid = threadIdx.x;
  int wave = tid >> 5, lane = tid & 31;
  int kh = lane >> 4, rl = lane & 15;
  v8f acc[4] = {};
  int steps = K >> 5;

  auto compute = [&](const __bf16* Ab, const __bf16* Bb) {
    RegTile a;
    #pragma unroll
    for (int j = 0; j < 8; ++j)
      a.u[j] = *(const unsigned int*)(Ab + (wave * 16 + rl) * 32 + AK(j, kh));
    #pragma unroll
    for (int s = 0; s < 4; ++s) {
      RegTile b;
      #pragma unroll
      for (int j = 0; j < 8; ++j)
        b.u[j] = *(const unsigned int*)(Bb + (s * 16 + rl) * 32 + AK(j, kh));
      acc[s] = __builtin_amdgcn_wmma_f32_16x16x32_bf16(false, a.v, false, b.v,
                                                       (short)0, acc[s], false, false);
    }
  };

#if HAVE_ASYNC_LDS
  int rowA = tid >> 2, seg = (tid & 3) * 8;   // 16B chunks
  auto issue = [&](int buf, int k0) {
    __builtin_amdgcn_global_load_async_to_lds_b128(
        (as1_v4i*)(A + (size_t)(m0 + rowA) * K + k0 + seg),
        (as3_v4i*)(&As[buf][rowA * 32 + seg]), 0, 0);
    __builtin_amdgcn_global_load_async_to_lds_b128(
        (as1_v4i*)(A + (size_t)(m0 + rowA + 64) * K + k0 + seg),
        (as3_v4i*)(&As[buf][(rowA + 64) * 32 + seg]), 0, 0);
    __builtin_amdgcn_global_load_async_to_lds_b128(
        (as1_v4i*)(BT + (size_t)(n0 + rowA) * K + k0 + seg),
        (as3_v4i*)(&Bs[buf][rowA * 32 + seg]), 0, 0);
  };
  issue(0, 0);
  for (int i = 0; i < steps; ++i) {
    int buf = i & 1;
    __builtin_amdgcn_s_wait_asynccnt(0);   // stage i landed in LDS (this wave)
    __syncthreads();                       // ... and for every wave
    if (i + 1 < steps) issue(buf ^ 1, (i + 1) * 32);
    if (i + 2 < steps)
      __builtin_prefetch(A + (size_t)(m0 + rowA) * K + (i + 2) * 32, 0, 1);
    compute(As[buf], Bs[buf]);
  }
#else
  for (int i = 0; i < steps; ++i) {
    int k0 = i * 32;
    __syncthreads();
    {
      int row = tid >> 1, half = (tid & 1) * 16;
      *(uint4*)(&As[0][row * 32 + half]) =
          *(const uint4*)(A + (size_t)(m0 + row) * K + k0 + half);
    }
    {
      int n = tid >> 2, seg = (tid & 3) * 8;
      *(uint4*)(&Bs[0][n * 32 + seg]) =
          *(const uint4*)(BT + (size_t)(n0 + n) * K + k0 + seg);
    }
    __syncthreads();
    compute(As[0], Bs[0]);
  }
#endif
  #pragma unroll
  for (int s = 0; s < 4; ++s)
    #pragma unroll
    for (int r = 0; r < 8; ++r) {
      int m = m0 + wave * 16 + r + 8 * kh;
      C[(size_t)m * N + n0 + s * 16 + rl] = acc[s][r];
    }
}

// -------------------- qkv epilogue: bias + per-head scatter ----------------
__global__ __launch_bounds__(256) void eqkv_kernel(const float* __restrict__ C,
                                                   const float* __restrict__ bias,
                                                   __bf16* __restrict__ qkvh) {
  int idx = blockIdx.x * 256 + threadIdx.x;   // NT * 576
  if (idx >= NT * 576) return;
  int t = idx / 576, j = idx - t * 576;
  int sh = j >> 5, d = j & 31;                // sh = s*6 + h (0..17)
  int g = t >> 8, l = t & 255;
  float v = 0.f;
  if (d < HD) { int col = sh * HD + d; v = C[(size_t)t * QKV_NP + col] + bias[col]; }
  qkvh[((size_t)(g * 18 + sh) * 256 + l) * HDP + d] = (__bf16)v;
}

// ------------------------- attention per (window, head) --------------------
__global__ __launch_bounds__(256) void attn_kernel(const __bf16* __restrict__ qkvh,
                                                   __bf16* __restrict__ av) {
  extern __shared__ char smem[];
  __bf16* qs  = (__bf16*)smem;      // [256][32]
  __bf16* ks  = qs + 8192;          // [256][32]
  __bf16* vsT = ks + 8192;          // [32][256]  (transposed v)
  __bf16* ps  = vsT + 8192;         // per-wave [16][256] bf16 strips
  const float scale = 0.18257418583505536f;  // 1/sqrt(30)
  int g = blockIdx.x / NHEAD, h = blockIdx.x % NHEAD;
  int tid = threadIdx.x, wave = tid >> 5, lane = tid & 31;
  int kh = lane >> 4, rl = lane & 15;
  size_t baseQ = (size_t)(g * 18 + h) * 8192;
  size_t baseK = (size_t)(g * 18 + 6 + h) * 8192;
  size_t baseV = (size_t)(g * 18 + 12 + h) * 8192;
#if HAVE_ASYNC_LDS
  for (int i = tid; i < 1024; i += 256) {
    __builtin_amdgcn_global_load_async_to_lds_b128(
        (as1_v4i*)(qkvh + baseQ + i * 8), (as3_v4i*)(qs + i * 8), 0, 0);
    __builtin_amdgcn_global_load_async_to_lds_b128(
        (as1_v4i*)(qkvh + baseK + i * 8), (as3_v4i*)(ks + i * 8), 0, 0);
  }
#else
  for (int i = tid; i < 1024; i += 256) {
    ((uint4*)qs)[i] = ((const uint4*)(qkvh + baseQ))[i];
    ((uint4*)ks)[i] = ((const uint4*)(qkvh + baseK))[i];
  }
#endif
  for (int i = tid; i < 8192; i += 256)
    vsT[(i & 31) * 256 + (i >> 5)] = qkvh[baseV + i];
#if HAVE_ASYNC_LDS
  __builtin_amdgcn_s_wait_asynccnt(0);
#endif
  __syncthreads();
  __bf16* pw = ps + wave * 16 * 256;
  for (int strip = 0; strip < 2; ++strip) {
    int m0 = (wave + strip * 8) * 16;
    v8f acc[16] = {};
    RegTile a;
    #pragma unroll
    for (int j = 0; j < 8; ++j)
      a.u[j] = *(const unsigned int*)(qs + (m0 + rl) * 32 + AK(j, kh));
    #pragma unroll
    for (int t = 0; t < 16; ++t) {       // scores = q . k^T   (K = 32 channels)
      RegTile b;
      #pragma unroll
      for (int j = 0; j < 8; ++j)
        b.u[j] = *(const unsigned int*)(ks + (t * 16 + rl) * 32 + AK(j, kh));
      acc[t] = __builtin_amdgcn_wmma_f32_16x16x32_bf16(false, a.v, false, b.v,
                                                       (short)0, acc[t], false, false);
    }
    #pragma unroll
    for (int r = 0; r < 8; ++r) {        // row softmax (half-wave shuffles)
      float mx = -3.0e38f;
      #pragma unroll
      for (int t = 0; t < 16; ++t) mx = fmaxf(mx, acc[t][r]);
      #pragma unroll
      for (int s = 1; s < 16; s <<= 1) mx = fmaxf(mx, __shfl_xor(mx, s, 32));
      float sum = 0.f;
      #pragma unroll
      for (int t = 0; t < 16; ++t) {
        float e = __expf((acc[t][r] - mx) * scale);
        acc[t][r] = e;  sum += e;
      }
      #pragma unroll
      for (int s = 1; s < 16; s <<= 1) sum += __shfl_xor(sum, s, 32);
      float inv = 1.f / sum;
      int row = r + 8 * kh;
      #pragma unroll
      for (int t = 0; t < 16; ++t)
        pw[row * 256 + t * 16 + rl] = (__bf16)(acc[t][r] * inv);
    }
    v8f o0 = {}, o1 = {};                // av = P . V   (K = 256 tokens)
    for (int k0 = 0; k0 < 256; k0 += 32) {
      RegTile pa, b0, b1;
      #pragma unroll
      for (int j = 0; j < 8; ++j) {
        pa.u[j] = *(const unsigned int*)(pw + rl * 256 + k0 + AK(j, kh));
        b0.u[j] = *(const unsigned int*)(vsT + rl * 256        + k0 + AK(j, kh));
        b1.u[j] = *(const unsigned int*)(vsT + (16 + rl) * 256 + k0 + AK(j, kh));
      }
      o0 = __builtin_amdgcn_wmma_f32_16x16x32_bf16(false, pa.v, false, b0.v, (short)0, o0, false, false);
      o1 = __builtin_amdgcn_wmma_f32_16x16x32_bf16(false, pa.v, false, b1.v, (short)0, o1, false, false);
    }
    #pragma unroll
    for (int r = 0; r < 8; ++r) {
      int tokloc = m0 + r + 8 * kh;
      size_t rowbase = ((size_t)g * 256 + tokloc) * EPAD + h * HD;
      av[rowbase + rl] = (__bf16)o0[r];
      if (16 + rl < HD) av[rowbase + 16 + rl] = (__bf16)o1[r];
    }
  }
}

// --------------------------- elementwise epilogues -------------------------
__global__ __launch_bounds__(256) void emlp1_kernel(const float* __restrict__ C,
                                                    const float* __restrict__ bias,
                                                    __bf16* __restrict__ out) {
  int idx = blockIdx.x * 256 + threadIdx.x;    // NT * MLP_HP
  if (idx >= NT * MLP_HP) return;
  int t = idx / MLP_HP, j = idx - t * MLP_HP;
  float v = 0.f;
  if (j < MLP_H) v = gelu_tanh(C[(size_t)t * MLP_HP + j] + bias[j]);
  out[idx] = (__bf16)v;
}

__global__ __launch_bounds__(256) void eres_kernel(const float* __restrict__ C,
                                                   const float* __restrict__ bias,
                                                   float* __restrict__ tok, int ldc) {
  int idx = blockIdx.x * 256 + threadIdx.x;    // NT * EMB
  if (idx >= NT * EMB) return;
  int t = idx / EMB, j = idx - t * EMB;
  tok[idx] += C[(size_t)t * ldc + j] + bias[j];
}

// ----------- final: pixel-shuffle up + bicubic + gumbel-hard blend ---------
__global__ __launch_bounds__(256) void final_kernel(const float* __restrict__ x,
                                                    const float* __restrict__ upC,
                                                    const float* __restrict__ up_b,
                                                    const float* __restrict__ mask,
                                                    float* __restrict__ sr) {
  int idx = blockIdx.x * 256 + threadIdx.x;    // 3 * 1024 * 1024
  if (idx >= 3 * 1024 * 1024) return;
  int c = idx >> 20, rem = idx & 1048575;
  int Y = rem >> 10, X = rem & 1023;
  int ty = Y >> 8, tx = X >> 8, tile = ty * 4 + tx;
  int yy = Y & 255, xx = X & 255;
  int y = yy >> 2, xq = xx >> 2;
  int wy = y >> 4, wx = xq >> 4, ly = y & 15, lx = xq & 15;
  int t = ((tile * 16 + wy * 4 + wx) << 8) + ly * 16 + lx;
  int ucol = ((yy & 3) * 4 + (xx & 3)) * 3 + c;
  float uval = upC[(size_t)t * UP_NP + ucol] + up_b[ucol];
  float iy = (yy + 0.5f) * 0.25f - 0.5f;
  float ix = (xx + 0.5f) * 0.25f - 0.5f;
  int by = (int)floorf(iy), bx = (int)floorf(ix);
  float fy = iy - by, fx = ix - bx;
  float wyv[4], wxv[4];
  #pragma unroll
  for (int i = 0; i < 4; ++i) { wyv[i] = cubw(fy + 1.f - i); wxv[i] = cubw(fx + 1.f - i); }
  float neg = 0.f;
  #pragma unroll
  for (int i = 0; i < 4; ++i) {
    int sy = min(max(by - 1 + i, 0), 63);
    float rowacc = 0.f;
    #pragma unroll
    for (int j = 0; j < 4; ++j) {
      int sx = min(max(bx - 1 + j, 0), 63);
      rowacc += wxv[j] * x[c * 65536 + (ty * 64 + sy) * 256 + tx * 64 + sx];
    }
    neg += wyv[i] * rowacc;
  }
  float m = mask[tile];
  sr[idx] = m * uval + (1.f - m) * neg;
}

// ---------------------------------------------------------------------------
extern "C" void kernel_launch(void* const* d_in, const int* in_sizes, int n_in,
                              void* d_out, int out_size, void* d_ws, size_t ws_size,
                              hipStream_t stream) {
  const float* x      = (const float*)d_in[0];
  const float* gu     = (const float*)d_in[1];
  const float* sel1_w = (const float*)d_in[2];
  const float* sel1_b = (const float*)d_in[3];
  const float* sel2_w = (const float*)d_in[4];
  const float* sel2_b = (const float*)d_in[5];
  const float* sel3_w = (const float*)d_in[6];
  const float* sel3_b = (const float*)d_in[7];
  const float* embed_w= (const float*)d_in[8];
  const float* embed_b= (const float*)d_in[9];
  const float* ln1_g  = (const float*)d_in[10];
  const float* ln1_b  = (const float*)d_in[11];
  const float* qkv_w  = (const float*)d_in[12];
  const float* qkv_b  = (const float*)d_in[13];
  const float* proj_w = (const float*)d_in[14];
  const float* proj_b = (const float*)d_in[15];
  const float* ln2_g  = (const float*)d_in[16];
  const float* ln2_b  = (const float*)d_in[17];
  const float* mlp1_w = (const float*)d_in[18];
  const float* mlp1_b = (const float*)d_in[19];
  const float* mlp2_w = (const float*)d_in[20];
  const float* mlp2_b = (const float*)d_in[21];
  const float* up_w   = (const float*)d_in[22];
  const float* up_b   = (const float*)d_in[23];
  float* out = (float*)d_out;

  char* ws = (char*)d_ws;
  size_t off = 0;
  auto carve = [&](size_t bytes) -> char* {
    char* p = ws + off;
    off = (off + bytes + 255) & ~(size_t)255;
    return p;
  };
  __bf16* w_qkv  = (__bf16*)carve((size_t)EPAD * QKV_NP * 2);   // transposed [N][K]
  __bf16* w_proj = (__bf16*)carve((size_t)EPAD * EPAD * 2);
  __bf16* w_mlp1 = (__bf16*)carve((size_t)EPAD * MLP_HP * 2);
  __bf16* w_mlp2 = (__bf16*)carve((size_t)MLP_HP * EPAD * 2);
  __bf16* w_up   = (__bf16*)carve((size_t)EPAD * UP_NP * 2);
  float*  maskb  = (float*)carve(16 * 4);
  float*  tok    = (float*)carve((size_t)NT * EMB * 4);
  __bf16* hb     = (__bf16*)carve((size_t)NT * EPAD * 2);
  __bf16* qkvh   = (__bf16*)carve((size_t)NWIN * 18 * 256 * HDP * 2);
  __bf16* avb    = (__bf16*)carve((size_t)NT * EPAD * 2);
  __bf16* gel    = (__bf16*)carve((size_t)NT * MLP_HP * 2);
  float*  Cscr   = (float*)carve((size_t)NT * QKV_NP * 4);

  float* out_sr = out;
  float* out_f3 = out + 3145728;
  float* out_f2 = out + 3145760;
  float* out_f1 = out + 3146272;

  // ---- weights -> padded, transposed bf16 ----
  auto cvlaunch = [&](const float* s, __bf16* d, int K, int N, int Kp, int Np) {
    int tot = Np * Kp;
    convert_padT_kernel<<<(tot + 255) / 256, 256, 0, stream>>>(s, d, K, N, Kp, Np);
  };
  cvlaunch(qkv_w,  w_qkv,  180, 540, EPAD, QKV_NP);
  cvlaunch(proj_w, w_proj, 180, 180, EPAD, EPAD);
  cvlaunch(mlp1_w, w_mlp1, 180, 360, EPAD, MLP_HP);
  cvlaunch(mlp2_w, w_mlp2, 360, 180, MLP_HP, EPAD);
  cvlaunch(up_w,   w_up,   180, 48,  EPAD, UP_NP);

  // ---- selector MLP + gumbel hard mask ----
  sel_kernel<<<16, 64, 0, stream>>>(x, gu, sel1_w, sel1_b, sel2_w, sel2_b,
                                    sel3_w, sel3_b, maskb, out_f3, out_f2, out_f1);

  // ---- embed, LN1 ----
  embed_kernel<<<(NT * EMB) / 256, 256, 0, stream>>>(x, embed_w, embed_b, tok);
  ln_kernel<<<NT, 256, 0, stream>>>(tok, hb, ln1_g, ln1_b, 1);

  // ---- qkv GEMM + scatter ----
  gemm_bf16_kernel<<<dim3(NT / 128, QKV_NP / 64), 256, 0, stream>>>(hb, w_qkv, Cscr, NT, QKV_NP, EPAD);
  eqkv_kernel<<<(NT * 576) / 256, 256, 0, stream>>>(Cscr, qkv_b, qkvh);

  // ---- attention (one block per window x head, all-LDS) ----
  (void)hipMemsetAsync(avb, 0, (size_t)NT * EPAD * 2, stream);
  attn_kernel<<<NWIN * NHEAD, 256, 114688, stream>>>(qkvh, avb);

  // ---- proj + residual, LN2 ----
  gemm_bf16_kernel<<<dim3(NT / 128, EPAD / 64), 256, 0, stream>>>(avb, w_proj, Cscr, NT, EPAD, EPAD);
  eres_kernel<<<(NT * EMB) / 256, 256, 0, stream>>>(Cscr, proj_b, tok, EPAD);
  ln_kernel<<<NT, 256, 0, stream>>>(tok, hb, ln2_g, ln2_b, 1);

  // ---- MLP ----
  gemm_bf16_kernel<<<dim3(NT / 128, MLP_HP / 64), 256, 0, stream>>>(hb, w_mlp1, Cscr, NT, MLP_HP, EPAD);
  emlp1_kernel<<<(NT * MLP_HP) / 256, 256, 0, stream>>>(Cscr, mlp1_b, gel);
  gemm_bf16_kernel<<<dim3(NT / 128, EPAD / 64), 256, 0, stream>>>(gel, w_mlp2, Cscr, NT, EPAD, MLP_HP);
  eres_kernel<<<(NT * EMB) / 256, 256, 0, stream>>>(Cscr, mlp2_b, tok, EPAD);

  // ---- up-projection ----
  ln_kernel<<<NT, 256, 0, stream>>>(tok, hb, ln1_g, ln1_b, 0);  // plain f32->bf16 pad
  gemm_bf16_kernel<<<dim3(NT / 128, UP_NP / 64), 256, 0, stream>>>(hb, w_up, Cscr, NT, UP_NP, EPAD);

  // ---- pixel shuffle + bicubic + blend ----
  final_kernel<<<(3 * 1024 * 1024) / 256, 256, 0, stream>>>(x, Cscr, up_b, maskb, out_sr);
}
